// Actor_16097537425467
// MI455X (gfx1250) — compile-verified, matrix operand order checked by reading, not run on hardware
//
#include <hip/hip_runtime.h>
#include <math.h>

typedef __attribute__((ext_vector_type(16))) _Float16 v16h;
typedef __attribute__((ext_vector_type(8)))  float    v8f;
typedef __attribute__((ext_vector_type(8)))  _Float16 h8;

#define WMMA_F16(A, B, C) \
  __builtin_amdgcn_wmma_f32_16x16x32_f16(false, (A), false, (B), (short)0, (C), false, false)

__device__ __forceinline__ float leaky01(float x) { return x > 0.f ? x : 0.01f * x; }
__device__ __forceinline__ float relu_(float x)   { return x > 0.f ? x : 0.f; }

// Build a 16-half A/B fragment from an f32 row pointer already offset to
// (row_base + kstep*32 + half*8). Elements 0..7 <- r[0..7], 8..15 <- r[16..23]
// per the ISA 16-bit 16x32 A-matrix VGPR layout.
__device__ __forceinline__ v16h frag_from_f32(const float* __restrict__ r) {
  const float4 x0 = *(const float4*)(r);
  const float4 x1 = *(const float4*)(r + 4);
  const float4 y0 = *(const float4*)(r + 16);
  const float4 y1 = *(const float4*)(r + 20);
  v16h f;
  f[0]  = (_Float16)x0.x; f[1]  = (_Float16)x0.y; f[2]  = (_Float16)x0.z; f[3]  = (_Float16)x0.w;
  f[4]  = (_Float16)x1.x; f[5]  = (_Float16)x1.y; f[6]  = (_Float16)x1.z; f[7]  = (_Float16)x1.w;
  f[8]  = (_Float16)y0.x; f[9]  = (_Float16)y0.y; f[10] = (_Float16)y0.z; f[11] = (_Float16)y0.w;
  f[12] = (_Float16)y1.x; f[13] = (_Float16)y1.y; f[14] = (_Float16)y1.z; f[15] = (_Float16)y1.w;
  return f;
}

// ---------------------------------------------------------------------------
// Kernel 1: wg = gs @ W_w^T + W_b ; g_term = wg.a_g ; s0 = leaky(g_term + wg.a_l + att_b)
// one thread per batch row (B = 4096 = 16 blocks * 256 threads)
// ---------------------------------------------------------------------------
__global__ void k1_global_proj(const float* __restrict__ gs,
                               const float* __restrict__ Ww,
                               const float* __restrict__ Wb,
                               const float* __restrict__ attw,
                               const float* __restrict__ attb,
                               float* __restrict__ wg_out,
                               float* __restrict__ gt_out,
                               float* __restrict__ s0_out) {
  __shared__ float sW[32 * 64];
  __shared__ float sWb[32], sag[32], sal[32];
  const int t = threadIdx.x;
  for (int i = t; i < 32 * 64; i += 256) sW[i] = Ww[i];
  if (t < 32) { sWb[t] = Wb[t]; sag[t] = attw[t]; sal[t] = attw[32 + t]; }
  __syncthreads();

  const int b = blockIdx.x * 256 + t;
  float g[64];
#pragma unroll
  for (int d = 0; d < 64; ++d) g[d] = gs[(size_t)b * 64 + d];

  float gt = 0.f, lt = 0.f;
  for (int h = 0; h < 32; ++h) {
    float s = sWb[h];
#pragma unroll
    for (int d = 0; d < 64; ++d) s += g[d] * sW[h * 64 + d];
    wg_out[(size_t)b * 32 + h] = s;
    gt += s * sag[h];
    lt += s * sal[h];
  }
  gt_out[b] = gt;
  s0_out[b] = leaky01(gt + lt + attb[0]);
}

// ---------------------------------------------------------------------------
// Kernel 2: fused attention pass over L. One wave per batch, 4 batches/wave.
// Per 16-row L chunk, 6 WMMAs:
//   c0/c1 : ul tile (16 x 32), C preloaded with U_b
//   c2    : score dots -- B-frag holds u_eff = U_w^T a_l broadcast to every
//           column, C preloaded with (g_term + a_l.U_b + att_b), so every
//           lane holds score-ready dot[M = v+8*half] in c2[v].  No shuffles,
//           no masking in the 12 full chunks; the 8-row tail chunk is peeled.
// ---------------------------------------------------------------------------
__global__ void k2_attention(const float* __restrict__ local,
                             const float* __restrict__ Uw,
                             const float* __restrict__ Ub,
                             const float* __restrict__ attw,
                             const float* __restrict__ attb,
                             const float* __restrict__ wg_in,
                             const float* __restrict__ gt_in,
                             const float* __restrict__ s0_in,
                             float* __restrict__ st_out) {
  const int lane = threadIdx.x & 31;
  const int wave = threadIdx.x >> 5;
  const int n    = lane & 15;
  const int half = lane >> 4;
  const int kb0  = half * 8;
  const int gw   = blockIdx.x * 8 + wave;   // 0..1023

  // u_eff[k] = sum_h a_l[h] * U_w[h][k] at this lane's 32 K slots -> B frags
  float cub = 0.f;
  v16h Bu0, Bu1;
  {
    float ue[32];
#pragma unroll
    for (int i = 0; i < 32; ++i) ue[i] = 0.f;
    for (int h = 0; h < 32; ++h) {
      const float al = attw[32 + h];
      const float* ur = Uw + h * 64;
#pragma unroll
      for (int i = 0; i < 8; ++i) {
        ue[i]      += al * ur[kb0 + i];
        ue[8 + i]  += al * ur[kb0 + 16 + i];
        ue[16 + i] += al * ur[kb0 + 32 + i];
        ue[24 + i] += al * ur[kb0 + 48 + i];
      }
      cub += al * Ub[h];
    }
#pragma unroll
    for (int j = 0; j < 16; ++j) { Bu0[j] = (_Float16)ue[j]; Bu1[j] = (_Float16)ue[16 + j]; }
  }
  const float attb0 = attb[0];
  const float ub0 = Ub[n], ub1 = Ub[16 + n];

  // B fragments of U_w (N-tile 0: h 0..15, N-tile 1: h 16..31; K-steps 0/1)
  v16h Bf00, Bf01, Bf10, Bf11;
  {
    const float* r0 = Uw + (size_t)n * 64;
    const float* r1 = Uw + (size_t)(16 + n) * 64;
    Bf00 = frag_from_f32(r0 + kb0);
    Bf10 = frag_from_f32(r0 + 32 + kb0);
    Bf01 = frag_from_f32(r1 + kb0);
    Bf11 = frag_from_f32(r1 + 32 + kb0);
  }

  for (int k = 0; k < 4; ++k) {
    const int b = gw * 4 + k;
    const float s0b    = s0_in[b];
    const float wgl    = wg_in[(size_t)b * 32 + lane];
    const float c2init = gt_in[b] + cub + attb0;   // score = leaky(dot + c2init)
    float Sacc = 0.f, V0 = 0.f, V1 = 0.f;
    const float* base = local + ((size_t)b * 200 + n) * 64;

    // ---- 12 full 16-row chunks (rows 0..191), no masking ----
    for (int c = 0; c < 12; ++c) {
      const float* row = base + (size_t)c * 16 * 64;
      __builtin_prefetch(row + 16 * 64, 0, 0);   // next chunk -> global_prefetch_b8

      const v16h a0 = frag_from_f32(row + kb0);
      const v16h a1 = frag_from_f32(row + 32 + kb0);

      v8f c2, c0, c1;
#pragma unroll
      for (int i = 0; i < 8; ++i) { c2[i] = c2init; c0[i] = ub0; c1[i] = ub1; }
      // scores first so the c2 -> VALU hazard is covered by the c0/c1 WMMAs
      c2 = WMMA_F16(a0, Bu0,  c2);
      c2 = WMMA_F16(a1, Bu1,  c2);
      c0 = WMMA_F16(a0, Bf00, c0);
      c0 = WMMA_F16(a1, Bf10, c0);
      c1 = WMMA_F16(a0, Bf01, c1);
      c1 = WMMA_F16(a1, Bf11, c1);

#pragma unroll
      for (int v = 0; v < 8; ++v) {
        const float sc = leaky01(c2[v]);
        Sacc += sc;
        V0 += sc * c0[v];
        V1 += sc * c1[v];
      }
    }

    // ---- tail chunk: rows 192..199 (8 valid rows -> only half 0 counts) ----
    {
      const float* row = base + (size_t)192 * 64 - (size_t)(n >= 8 ? (n - 7) : 0) * 64;
      // rows 192+n for n<8; clamp n>=8 into range (masked out below anyway)
      const v16h a0 = frag_from_f32(row + kb0);
      const v16h a1 = frag_from_f32(row + 32 + kb0);

      v8f c2, c0, c1;
#pragma unroll
      for (int i = 0; i < 8; ++i) { c2[i] = c2init; c0[i] = ub0; c1[i] = ub1; }
      c2 = WMMA_F16(a0, Bu0,  c2);
      c2 = WMMA_F16(a1, Bu1,  c2);
      c0 = WMMA_F16(a0, Bf00, c0);
      c0 = WMMA_F16(a1, Bf10, c0);
      c1 = WMMA_F16(a0, Bf01, c1);
      c1 = WMMA_F16(a1, Bf11, c1);

      if (half == 0) {                 // M = v (0..7) valid; M = 8..15 padded
#pragma unroll
        for (int v = 0; v < 8; ++v) {
          const float sc = leaky01(c2[v]);
          Sacc += sc;
          V0 += sc * c0[v];
          V1 += sc * c1[v];
        }
      }
    }

    // Sacc is uniform within each 16-lane half; combine the two halves.
    const float S   = Sacc + __shfl_xor(Sacc, 16, 32);
    const float inv = 1.0f / (s0b + S);
    const float Vf0 = V0 + __shfl_xor(V0, 16, 32);   // V[h = n]
    const float Vf1 = V1 + __shfl_xor(V1, 16, 32);   // V[h = 16+n]

    float* st = st_out + (size_t)b * 64;
    st[lane] = relu_(s0b * inv * wgl);               // n0 * wg
    if (half == 0) st[32 + n] = relu_(Vf0 * inv);    // agg_local[0..15]
    else           st[48 + n] = relu_(Vf1 * inv);    // agg_local[16..31]
  }
}

// ---------------------------------------------------------------------------
// Kernel 3: MLP 64 -> 256 -> 256 -> 8 (relu, relu, sigmoid), WMMA tiles.
// One wave per 16-batch tile; 256 tiles = 32 blocks * 8 waves.
// Activations round-trip through an 8KB/wave LDS tile (C layout -> A layout).
// ---------------------------------------------------------------------------
__global__ void k3_mlp(const float* __restrict__ st,
                       const float* __restrict__ l1w, const float* __restrict__ l1b,
                       const float* __restrict__ l2w, const float* __restrict__ l2b,
                       const float* __restrict__ l3w, const float* __restrict__ l3b,
                       float* __restrict__ out) {
  extern __shared__ _Float16 smem[];
  const int lane = threadIdx.x & 31;
  const int wave = threadIdx.x >> 5;
  const int n    = lane & 15;
  const int half = lane >> 4;
  _Float16* act = smem + (size_t)wave * 4096;        // 16 x 256 f16 per wave
  const int tile = blockIdx.x * 8 + wave;            // 0..255
  const int bb   = tile * 16 + n;                    // this lane's A row (batch)

  // A fragments of states (K = 64 -> 2 K-steps)
  v16h a1[2];
#pragma unroll
  for (int ks = 0; ks < 2; ++ks)
    a1[ks] = frag_from_f32(st + (size_t)bb * 64 + ks * 32 + half * 8);

  // ---- layer 1: K=64, N=256 ----
  for (int nt = 0; nt < 16; ++nt) {
    const float bias = l1b[nt * 16 + n];
    v8f acc;
#pragma unroll
    for (int i = 0; i < 8; ++i) acc[i] = bias;
#pragma unroll
    for (int ks = 0; ks < 2; ++ks) {
      v16h bf = frag_from_f32(l1w + (size_t)(nt * 16 + n) * 64 + ks * 32 + half * 8);
      acc = WMMA_F16(a1[ks], bf, acc);
    }
#pragma unroll
    for (int v = 0; v < 8; ++v)
      act[(v + 8 * half) * 256 + nt * 16 + n] = (_Float16)relu_(acc[v]);
  }
  __syncthreads();

  v16h a2[8];
#pragma unroll
  for (int ks = 0; ks < 8; ++ks) {
    const _Float16* r = act + n * 256 + ks * 32 + half * 8;
    h8 lo = *(const h8*)r;
    h8 hi = *(const h8*)(r + 16);
#pragma unroll
    for (int i = 0; i < 8; ++i) { a2[ks][i] = lo[i]; a2[ks][8 + i] = hi[i]; }
  }
  __syncthreads();

  // ---- layer 2: K=256, N=256 ----
  for (int nt = 0; nt < 16; ++nt) {
    const float bias = l2b[nt * 16 + n];
    v8f acc;
#pragma unroll
    for (int i = 0; i < 8; ++i) acc[i] = bias;
#pragma unroll
    for (int ks = 0; ks < 8; ++ks) {
      v16h bf = frag_from_f32(l2w + (size_t)(nt * 16 + n) * 256 + ks * 32 + half * 8);
      acc = WMMA_F16(a2[ks], bf, acc);
    }
#pragma unroll
    for (int v = 0; v < 8; ++v)
      act[(v + 8 * half) * 256 + nt * 16 + n] = (_Float16)relu_(acc[v]);
  }
  __syncthreads();

  v16h a3[8];
#pragma unroll
  for (int ks = 0; ks < 8; ++ks) {
    const _Float16* r = act + n * 256 + ks * 32 + half * 8;
    h8 lo = *(const h8*)r;
    h8 hi = *(const h8*)(r + 16);
#pragma unroll
    for (int i = 0; i < 8; ++i) { a3[ks][i] = lo[i]; a3[ks][8 + i] = hi[i]; }
  }

  // ---- layer 3: K=256, N=8 (padded to 16) ----
  v8f acc;
  {
    const float bias = (n < 8) ? l3b[n] : 0.f;
#pragma unroll
    for (int i = 0; i < 8; ++i) acc[i] = bias;
  }
#pragma unroll
  for (int ks = 0; ks < 8; ++ks) {
    v16h bf;
    if (n < 8) {
      bf = frag_from_f32(l3w + (size_t)n * 256 + ks * 32 + half * 8);
    } else {
#pragma unroll
      for (int i = 0; i < 16; ++i) bf[i] = (_Float16)0.f;
    }
    acc = WMMA_F16(a3[ks], bf, acc);
  }

  if (n < 8) {
#pragma unroll
    for (int v = 0; v < 8; ++v) {
      const int row = v + 8 * half;
      const int b   = tile * 16 + row;
      out[(size_t)b * 8 + n] = 1.0f / (1.0f + __expf(-acc[v]));  // MAX_ACTION = 1
    }
  }
}

// ---------------------------------------------------------------------------
extern "C" void kernel_launch(void* const* d_in, const int* in_sizes, int n_in,
                              void* d_out, int out_size, void* d_ws, size_t ws_size,
                              hipStream_t stream) {
  (void)in_sizes; (void)n_in; (void)out_size; (void)ws_size;
  const float* gs    = (const float*)d_in[0];
  const float* local = (const float*)d_in[1];
  const float* Ww    = (const float*)d_in[2];
  const float* Wb    = (const float*)d_in[3];
  const float* Uw    = (const float*)d_in[4];
  const float* Ub    = (const float*)d_in[5];
  const float* attw  = (const float*)d_in[6];
  const float* attb  = (const float*)d_in[7];
  const float* l1w   = (const float*)d_in[8];
  const float* l1b   = (const float*)d_in[9];
  const float* l2w   = (const float*)d_in[10];
  const float* l2b   = (const float*)d_in[11];
  const float* l3w   = (const float*)d_in[12];
  const float* l3b   = (const float*)d_in[13];
  float* out = (float*)d_out;

  float* ws       = (float*)d_ws;
  float* ws_wg    = ws;                       // 4096*32
  float* ws_gt    = ws + 131072;              // 4096
  float* ws_s0    = ws + 135168;              // 4096
  float* ws_state = ws + 139264;              // 4096*64

  k1_global_proj<<<16, 256, 0, stream>>>(gs, Ww, Wb, attw, attb, ws_wg, ws_gt, ws_s0);
  k2_attention<<<128, 256, 0, stream>>>(local, Uw, Ub, attw, attb,
                                        ws_wg, ws_gt, ws_s0, ws_state);
  k3_mlp<<<32, 256, 8 * 4096 * sizeof(_Float16), stream>>>(ws_state,
                                                           l1w, l1b, l2w, l2b, l3w, l3b, out);
}
